// EmBed_RCLSTM_56762287784592
// MI455X (gfx1250) — matrix operand order repaired and checked by reading, not compiled, not run on hardware
//
#include <hip/hip_runtime.h>

// ---------------------------------------------------------------------------
// MI455X (gfx1250, wave32) RC-LSTM.
//  - batch-partitioned recurrence: 16 blocks x 256 threads (8 waves); each
//    block owns 16 batch rows and runs all 256 timesteps, no global sync.
//  - per step: [16 x 1536] @ [1536 x 1024] via v_wmma_f32_16x16x32_bf16
//    (A from LDS, B = bf16 weights streamed from the 192MB L2), f32 accum.
//  - x_t tiles staged into double-buffered LDS by the Tensor Data Mover
//    (tensor_load_to_lds + s_wait_tensorcnt), prefetching t+1 during t.
//  - cell state c lives in VGPRs in WMMA C/D layout for the whole kernel.
// ---------------------------------------------------------------------------

typedef __attribute__((ext_vector_type(16))) __bf16          v16bf;
typedef __attribute__((ext_vector_type(8)))  float           v8f;
typedef __attribute__((ext_vector_type(8)))  unsigned short  v8us;
typedef __attribute__((ext_vector_type(16))) unsigned short  v16us;
typedef __attribute__((ext_vector_type(4)))  unsigned int    v4u;
typedef __attribute__((ext_vector_type(8)))  int             v8i;
typedef __attribute__((ext_vector_type(4)))  int             v4i;

#define T_STEPS 256
#define BATCH   256
#define DIM_I   512
#define DIM_H   1024
#define DIM_C   1536
#define HLP     1032   // hidden LDS pitch (ushorts): 16B-aligned, 4-bank rotate
#define XLP     520    // x LDS pitch (ushorts):      16B-aligned, 4-bank rotate

// f32 -> bf16 round-to-nearest-even (storage only)
__device__ __forceinline__ unsigned short f2bf(float f) {
  unsigned int u = __float_as_uint(f);
  u += 0x7FFFu + ((u >> 16) & 1u);
  return (unsigned short)(u >> 16);
}

// ---------------------------------------------------------------------------
// TDM: DMA a 16 x 512 bf16 tile (row stride 512) from global into LDS with
// 16-byte padding after every 1024-byte row (pad_interval=256 DW, amount=4 DW)
// so the LDS image has pitch XLP=520 ushorts.
// ---------------------------------------------------------------------------
__device__ __forceinline__ void tdm_load_x_tile(const unsigned short* src,
                                                unsigned lds_byte_addr) {
  unsigned long long ga = (unsigned long long)(uintptr_t)src;
  v4u g0;
  g0[0] = 1u;                                        // count=1, user mode
  g0[1] = lds_byte_addr;                             // lds_addr [63:32]
  g0[2] = (unsigned)(ga & 0xFFFFFFFFu);              // global_addr[31:0]
  g0[3] = (unsigned)((ga >> 32) & 0x01FFFFFFull)     // global_addr[56:32]
        | (2u << 30);                                // type=2 (image)
  v8i g1;
  g1[0] = (int)((1u << 16)      // data_size = 2 bytes
              | (1u << 20)      // pad_enable
              | (7u << 22)      // pad_interval: 256 DWORDs
              | (3u << 25));    // pad_amount:   4 DWORDs (16 bytes)
  g1[1] = (int)(512u << 16);    // tensor_dim0 = 512 (bits 79:48)
  g1[2] = (int)(16u << 16);     // tensor_dim1 = 16  (bits 111:80)
  g1[3] = (int)(512u << 16);    // tile_dim0 = 512   (bits 127:112)
  g1[4] = 16;                   // tile_dim1 = 16    (bits 143:128)
  g1[5] = 512;                  // tensor_dim0_stride = 512 (bits 207:160)
  g1[6] = 0;
  g1[7] = 0;
  v4i gz = {0, 0, 0, 0};
#if defined(__clang_major__) && (__clang_major__ >= 23)
  v8i gz8 = {0, 0, 0, 0, 0, 0, 0, 0};
  __builtin_amdgcn_tensor_load_to_lds(g0, g1, gz, gz, gz8, 0);
#else
  __builtin_amdgcn_tensor_load_to_lds(g0, g1, gz, gz, 0);
#endif
}

// ---------------------------------------------------------------------------
// Kernel 1a: convert W_gate [1024x1536] and W_out [512x1024] to bf16 in ws
// ---------------------------------------------------------------------------
__global__ __launch_bounds__(256) void rclstm_convert_w(
    const float* __restrict__ Wg, const float* __restrict__ Wof,
    unsigned short* __restrict__ Wc, unsigned short* __restrict__ Wo) {
  const int n1 = DIM_H * DIM_C;
  const int n2 = DIM_I * DIM_H;
  int idx = blockIdx.x * 256 + threadIdx.x;
  if (idx < n1)           Wc[idx]      = f2bf(Wg[idx]);
  else if (idx < n1 + n2) Wo[idx - n1] = f2bf(Wof[idx - n1]);
}

// Kernel 1b: convert x [256*256*512] f32 -> bf16 (feeds the TDM path)
__global__ __launch_bounds__(256) void rclstm_convert_x(
    const float* __restrict__ xf, unsigned short* __restrict__ xb) {
  size_t idx = (size_t)blockIdx.x * 256 + threadIdx.x;   // float4 index
  const size_t n4 = (size_t)T_STEPS * BATCH * DIM_I / 4;
  if (idx < n4) {
    float4 v = ((const float4*)xf)[idx];
    ushort4 o;
    o.x = f2bf(v.x); o.y = f2bf(v.y); o.z = f2bf(v.z); o.w = f2bf(v.w);
    ((ushort4*)xb)[idx] = o;
  }
}

// ---------------------------------------------------------------------------
// Kernel 2: sequential recurrence, batch-partitioned. USE_TDM selects the
// Tensor-Data-Mover x path (bf16 x in global) vs inline f32->bf16 staging.
// ---------------------------------------------------------------------------
template <bool USE_TDM>
__global__ __launch_bounds__(256) void rclstm_recurrence(
    const float* __restrict__ x32, const unsigned short* __restrict__ xbf,
    const unsigned short* __restrict__ Wc, const float* __restrict__ b_gate,
    unsigned short* __restrict__ hbuf) {
  __shared__ unsigned short hlds[16 * HLP];
  __shared__ unsigned short xlds[2 * 16 * XLP];

  const int tid  = threadIdx.x;
  const int lane = tid & 31;
  const int wv   = tid >> 5;
  const int half = lane >> 4;
  const int ln   = lane & 15;
  const int row0 = blockIdx.x * 16;
  const int nbase = wv * 128;

  // h0 = 0
  for (int f = tid; f < 16 * DIM_H; f += 256)
    hlds[(f >> 10) * HLP + (f & 1023)] = 0;

  float bias[8];
#pragma unroll
  for (int tn = 0; tn < 8; ++tn) bias[tn] = b_gate[nbase + tn * 16 + ln];

  v8f cst[8];
#pragma unroll
  for (int tn = 0; tn < 8; ++tn)
#pragma unroll
    for (int r = 0; r < 8; ++r) cst[tn][r] = 0.0f;

  if (USE_TDM && wv == 0) {  // prologue: DMA x(0) into buffer 0
    tdm_load_x_tile(xbf + ((size_t)0 * BATCH + row0) * DIM_I,
                    (unsigned)(uintptr_t)&xlds[0]);
  }

  for (int t = 0; t < T_STEPS; ++t) {
    unsigned short* xb = &xlds[(t & 1) * (16 * XLP)];

    if (USE_TDM) {
      if (wv == 0) {
        if (t + 1 < T_STEPS)  // prefetch x(t+1) into the other buffer
          tdm_load_x_tile(xbf + ((size_t)(t + 1) * BATCH + row0) * DIM_I,
                          (unsigned)(uintptr_t)&xlds[((t + 1) & 1) * (16 * XLP)]);
        if (t + 1 < T_STEPS) __builtin_amdgcn_s_wait_tensorcnt(1);  // x(t) done
        else                 __builtin_amdgcn_s_wait_tensorcnt(0);
      }
    } else {
      // inline staging: f32 -> bf16, coalesced float4
      const float4* xv = (const float4*)(x32 + ((size_t)t * BATCH + row0) * DIM_I);
      for (int f = tid; f < 16 * (DIM_I / 4); f += 256) {
        int m = f >> 7, i4 = f & 127;
        float4 v = xv[m * (DIM_I / 4) + i4];
        unsigned short* p = &xb[m * XLP + i4 * 4];
        p[0] = f2bf(v.x); p[1] = f2bf(v.y); p[2] = f2bf(v.z); p[3] = f2bf(v.w);
      }
    }
    __syncthreads();  // x(t) + h(t-1) visible before A-matrix reads

    v8f acc[8];
#pragma unroll
    for (int tn = 0; tn < 8; ++tn)
#pragma unroll
      for (int r = 0; r < 8; ++r) acc[tn][r] = bias[tn];

    // ---- phase H: K = 0..1024 over hidden state ----
    {
      const unsigned short* arow = &hlds[ln * HLP];
      for (int kb = 0; kb < DIM_H; kb += 32) {
        v8us lo = *(const v8us*)(arow + kb + half * 16);
        v8us hi = *(const v8us*)(arow + kb + half * 16 + 8);
        v16us araw = __builtin_shufflevector(lo, hi,
            0, 1, 2, 3, 4, 5, 6, 7, 8, 9, 10, 11, 12, 13, 14, 15);
        v16bf a = __builtin_bit_cast(v16bf, araw);
        const unsigned short* wk = Wc + kb + half * 16;
#pragma unroll
        for (int tn = 0; tn < 8; ++tn) {
          int j = nbase + tn * 16 + ln;
          v16bf b = __builtin_bit_cast(v16bf, *(const v16us*)(wk + (size_t)j * DIM_C));
          acc[tn] = __builtin_amdgcn_wmma_f32_16x16x32_bf16(
              false, a, false, b, (short)0, acc[tn], false, false);
        }
      }
    }
    // ---- phase X: K = 0..512 over x_t ----
    {
      const unsigned short* arow = &xb[ln * XLP];
      for (int kb = 0; kb < DIM_I; kb += 32) {
        v8us lo = *(const v8us*)(arow + kb + half * 16);
        v8us hi = *(const v8us*)(arow + kb + half * 16 + 8);
        v16us araw = __builtin_shufflevector(lo, hi,
            0, 1, 2, 3, 4, 5, 6, 7, 8, 9, 10, 11, 12, 13, 14, 15);
        v16bf a = __builtin_bit_cast(v16bf, araw);
        const unsigned short* wk = Wc + DIM_H + kb + half * 16;
#pragma unroll
        for (int tn = 0; tn < 8; ++tn) {
          int j = nbase + tn * 16 + ln;
          v16bf b = __builtin_bit_cast(v16bf, *(const v16us*)(wk + (size_t)j * DIM_C));
          acc[tn] = __builtin_amdgcn_wmma_f32_16x16x32_bf16(
              false, a, false, b, (short)0, acc[tn], false, false);
        }
      }
    }
    __syncthreads();  // all LDS reads done before h(t) overwrite

    const bool last = (t == T_STEPS - 1);
#pragma unroll
    for (int tn = 0; tn < 8; ++tn) {
      const int col = nbase + tn * 16 + ln;
#pragma unroll
      for (int r = 0; r < 8; ++r) {
        float pre = acc[tn][r];
        float eg  = __expf(-pre);
        float g   = 1.0f / (1.0f + eg);                 // sigmoid(pre)
        float tp  = 2.0f / (1.0f + eg * eg) - 1.0f;     // tanh(pre), same exp
        float c   = cst[tn][r];
        c = c * g + tp * g;
        cst[tn][r] = c;
        float ec  = __expf(-c);
        float tc  = 2.0f / (1.0f + ec * ec) - 1.0f;     // tanh(c)
        float h   = tc * g;
        unsigned short hb = f2bf(h);
        int m = r + half * 8;
        hlds[m * HLP + col] = hb;
        if (last) hbuf[(size_t)(row0 + m) * DIM_H + col] = hb;
      }
    }
    // next iteration's DMA/staging targets the other x buffer and hlds writes
    // are ordered against reads by the pre-compute __syncthreads.
  }
}

// ---------------------------------------------------------------------------
// Kernel 3: out = h_T @ W_out^T + b_out   ([256x1024] @ [1024x512])
// ---------------------------------------------------------------------------
__global__ __launch_bounds__(256) void rclstm_out_gemm(
    const unsigned short* __restrict__ hbuf, const unsigned short* __restrict__ Wo,
    const float* __restrict__ b_out, float* __restrict__ out) {
  const int tid  = threadIdx.x;
  const int lane = tid & 31;
  const int wv   = tid >> 5;
  const int half = lane >> 4;
  const int ln   = lane & 15;
  const int row0 = blockIdx.x * 16;
  const int nbase = wv * 64;

  v8f acc[4];
#pragma unroll
  for (int tn = 0; tn < 4; ++tn) {
    float b = b_out[nbase + tn * 16 + ln];
#pragma unroll
    for (int r = 0; r < 8; ++r) acc[tn][r] = b;
  }

  const unsigned short* arow = hbuf + (size_t)(row0 + ln) * DIM_H;
  for (int kb = 0; kb < DIM_H; kb += 32) {
    v8us lo = *(const v8us*)(arow + kb + half * 16);
    v8us hi = *(const v8us*)(arow + kb + half * 16 + 8);
    v16us araw = __builtin_shufflevector(lo, hi,
        0, 1, 2, 3, 4, 5, 6, 7, 8, 9, 10, 11, 12, 13, 14, 15);
    v16bf a = __builtin_bit_cast(v16bf, araw);
#pragma unroll
    for (int tn = 0; tn < 4; ++tn) {
      int j = nbase + tn * 16 + ln;
      v16bf b = __builtin_bit_cast(v16bf,
          *(const v16us*)(Wo + (size_t)j * DIM_H + kb + half * 16));
      acc[tn] = __builtin_amdgcn_wmma_f32_16x16x32_bf16(
          false, a, false, b, (short)0, acc[tn], false, false);
    }
  }

#pragma unroll
  for (int tn = 0; tn < 4; ++tn) {
    int col = nbase + tn * 16 + ln;
#pragma unroll
    for (int r = 0; r < 8; ++r) {
      int m = r + half * 8;
      out[(size_t)(row0 + m) * DIM_I + col] = acc[tn][r];
    }
  }
}

// ---------------------------------------------------------------------------
extern "C" void kernel_launch(void* const* d_in, const int* in_sizes, int n_in,
                              void* d_out, int out_size, void* d_ws, size_t ws_size,
                              hipStream_t stream) {
  const float* x      = (const float*)d_in[0];  // [256,256,512]
  const float* W_gate = (const float*)d_in[1];  // [1024,1536]
  const float* b_gate = (const float*)d_in[2];  // [1024]
  const float* W_out  = (const float*)d_in[3];  // [512,1024]
  const float* b_out  = (const float*)d_in[4];  // [512]

  char* ws = (char*)d_ws;
  const size_t wc_b = (size_t)DIM_H * DIM_C * 2;            // 3 MB
  const size_t wo_b = (size_t)DIM_I * DIM_H * 2;            // 1 MB
  const size_t hb_b = (size_t)BATCH * DIM_H * 2;            // 0.5 MB
  const size_t xb_b = (size_t)T_STEPS * BATCH * DIM_I * 2;  // 64 MB
  unsigned short* Wc   = (unsigned short*)ws;
  unsigned short* Wo   = (unsigned short*)(ws + wc_b);
  unsigned short* hbuf = (unsigned short*)(ws + wc_b + wo_b);
  unsigned short* xbf  = (unsigned short*)(ws + wc_b + wo_b + hb_b);

  const int totw = DIM_H * DIM_C + DIM_I * DIM_H;
  rclstm_convert_w<<<(totw + 255) / 256, 256, 0, stream>>>(W_gate, W_out, Wc, Wo);

  if (ws_size >= wc_b + wo_b + hb_b + xb_b) {
    const size_t n4 = (size_t)T_STEPS * BATCH * DIM_I / 4;
    rclstm_convert_x<<<(int)((n4 + 255) / 256), 256, 0, stream>>>(x, xbf);
    rclstm_recurrence<true><<<BATCH / 16, 256, 0, stream>>>(x, xbf, Wc, b_gate, hbuf);
  } else {
    rclstm_recurrence<false><<<BATCH / 16, 256, 0, stream>>>(x, xbf, Wc, b_gate, hbuf);
  }
  rclstm_out_gemm<<<BATCH / 16, 256, 0, stream>>>(hbuf, Wo, b_out, (float*)d_out);

  (void)in_sizes; (void)n_in; (void)out_size; (void)ws_size;
}
